// MultiHeadAttention_70050916598293
// MI455X (gfx1250) — matrix operand order repaired and checked by reading, not compiled, not run on hardware
//
#include <hip/hip_runtime.h>

// ---------------------------------------------------------------------------
// Multi-head attention block for MI455X (gfx1250), bf16 WMMA pipeline.
//   1) convert x / weights fp32 -> bf16
//   2) fused QKV projection GEMM (128x128 tiles, WMMA bf16) + per-head
//      RMSNorm + RoPE epilogue; V stored transposed per head
//   3) flash attention per (head, 128-query block), online softmax, WMMA for
//      both Q@K^T and P@V, analytic causal mask (skips 64MB mask reads)
//   4) output projection GEMM (128x128 tiles, WMMA bf16, fp32 out)
// Tile staging uses GLOBAL_LOAD_ASYNC_TO_LDS (ASYNCcnt) when the toolchain
// exposes the builtins, else falls back to vector loads + ds stores.
// ---------------------------------------------------------------------------

typedef unsigned short u16;
typedef __attribute__((ext_vector_type(16))) __bf16        v16bf;
typedef __attribute__((ext_vector_type(8)))  float         v8f;
typedef __attribute__((ext_vector_type(4)))  unsigned int  v4u;
typedef __attribute__((ext_vector_type(2)))  unsigned int  v2u;
typedef __attribute__((ext_vector_type(4)))  float         v4f;

union Frag { v16bf v; v4u u[2]; };

#define SEQ    4096
#define DMODEL 1024
#define NHEADC 16
#define HDIM   64

#if defined(__AMDGCN__) && __has_builtin(__builtin_amdgcn_global_load_async_to_lds_b128) && \
    __has_builtin(__builtin_amdgcn_s_wait_asynccnt)
#define HAVE_ASYNC_LDS 1
typedef int gv4i __attribute__((vector_size(16)));           // matches builtin param
typedef __attribute__((address_space(1))) gv4i as1_v4i;      // global int4
typedef __attribute__((address_space(3))) gv4i as3_v4i;      // LDS int4
#else
#define HAVE_ASYNC_LDS 0
#endif

// copy 16B global -> LDS (async DMA when available)
__device__ __forceinline__ void g2l_b128(const u16* g, u16* l) {
#if HAVE_ASYNC_LDS
    __builtin_amdgcn_global_load_async_to_lds_b128(
        (as1_v4i*)(unsigned long long)g,
        (as3_v4i*)(unsigned int)(unsigned long long)l, 0, 0);
#else
    *(v4u*)l = *(const v4u*)g;
#endif
}
__device__ __forceinline__ void g2l_fence() {
#if HAVE_ASYNC_LDS
    __builtin_amdgcn_s_wait_asynccnt(0);
#endif
}

__device__ __forceinline__ u16 f2bf(float f) {  // fp32 -> bf16, round-nearest-even
    unsigned int x = __float_as_uint(f);
    unsigned int r = x + 0x7FFFu + ((x >> 16) & 1u);
    return (u16)(r >> 16);
}

// ---------------------------------------------------------------------------
// 0) elementwise fp32 -> bf16 conversion
// ---------------------------------------------------------------------------
__global__ __launch_bounds__(256) void cvt_bf16_kernel(const float* __restrict__ src,
                                                       u16* __restrict__ dst, int n) {
    int i = (blockIdx.x * 256 + threadIdx.x) * 4;
    if (i + 3 < n) {
        v4f f = *(const v4f*)(src + i);
        alignas(8) u16 o[4] = { f2bf(f[0]), f2bf(f[1]), f2bf(f[2]), f2bf(f[3]) };
        *(v2u*)(dst + i) = *(const v2u*)o;
    } else {
        for (; i < n; ++i) dst[i] = f2bf(src[i]);
    }
}

// ---------------------------------------------------------------------------
// 1) QKV projection: y = x @ W^T  (M=4096, N=1024, K=1024), 128x128 tile.
//    Block = 128 threads = 4 waves; wave owns 64x64 quadrant = 4x4 WMMA tiles
//    (16 WMMA per 16 ds_load_b128 fragment loads).
//    z = 0:Q, 1:K (RMSNorm + RoPE epilogue), 2:V (transposed store).
//    N tile of 128 spans exactly 2 heads (hd = 64).
// ---------------------------------------------------------------------------
__global__ __launch_bounds__(128) void qkv_gemm_kernel(
    const u16* __restrict__ xb,                        // [S][D] bf16
    const u16* __restrict__ wqb, const u16* __restrict__ wkb,
    const u16* __restrict__ wvb,                       // [N][K] bf16
    const float* __restrict__ cosT, const float* __restrict__ sinT,  // [S][32]
    const float* __restrict__ qnw,  const float* __restrict__ knw,   // [64]
    u16* __restrict__ qb, u16* __restrict__ kb,        // [S][D] bf16
    u16* __restrict__ vtb)                             // [H][HD][S] bf16 (transposed)
{
    const int nt    = blockIdx.x;          // 0..7, covers heads 2*nt, 2*nt+1
    const int mBase = blockIdx.y * 128;
    const int z     = blockIdx.z;
    const u16* w = (z == 0) ? wqb : (z == 1) ? wkb : wvb;
    const int nBase = nt * 128;

    __shared__ u16   At[128][40];          // [m][k] (80B rows, 16B aligned)
    __shared__ u16   Bt[128][40];          // [n][k]
    __shared__ float Ct[128][132];         // fp32 result tile for epilogue

    const int t    = threadIdx.x;
    const int lane = t & 31;
    const int wave = t >> 5;
    const int half = lane >> 4;
    const int ln   = lane & 15;
    const int mq   = wave >> 1, nq = wave & 1;   // 64x64 quadrant

    v8f acc[4][4];
    for (int i = 0; i < 4; ++i) for (int j = 0; j < 4; ++j)
        for (int e = 0; e < 8; ++e) acc[i][j][e] = 0.0f;

    for (int k0 = 0; k0 < DMODEL; k0 += 32) {
        __syncthreads();
        { // stage tiles: one 32-element row per thread, 4 x 16B async copies each
            const u16* ga = xb + (size_t)(mBase + t) * DMODEL + k0;
            u16*       la = &At[t][0];
            const u16* gb = w + (size_t)(nBase + t) * DMODEL + k0;
            u16*       lb = &Bt[t][0];
            for (int c = 0; c < 32; c += 8) g2l_b128(ga + c, la + c);
            for (int c = 0; c < 32; c += 8) g2l_b128(gb + c, lb + c);
            if (k0 + 32 < DMODEL) {        // prefetch next K tile
                __builtin_prefetch(ga + 32, 0, 3);
                __builtin_prefetch(gb + 32, 0, 3);
            }
        }
        g2l_fence();
        __syncthreads();

        Frag afr[4], bfr[4];
        for (int tm = 0; tm < 4; ++tm) {   // A 16x32: lane=row, K chunks {8h,16+8h}
            int m = mq * 64 + tm * 16 + ln;
            afr[tm].u[0] = *(const v4u*)&At[m][half * 8];
            afr[tm].u[1] = *(const v4u*)&At[m][16 + half * 8];
        }
        for (int tn = 0; tn < 4; ++tn) {   // B 32x16: lane=col, K chunks {8h,16+8h}
            int n = nq * 64 + tn * 16 + ln;
            bfr[tn].u[0] = *(const v4u*)&Bt[n][half * 8];
            bfr[tn].u[1] = *(const v4u*)&Bt[n][16 + half * 8];
        }
        for (int tm = 0; tm < 4; ++tm)
            for (int tn = 0; tn < 4; ++tn)
                acc[tm][tn] = __builtin_amdgcn_wmma_f32_16x16x32_bf16(
                    false, afr[tm].v, false, bfr[tn].v, (short)0, acc[tm][tn], false, false);
    }

    // spill result tile to LDS for row-wise epilogue
    for (int tm = 0; tm < 4; ++tm)
        for (int tn = 0; tn < 4; ++tn) {
            int col = nq * 64 + tn * 16 + ln;
            for (int v = 0; v < 8; ++v)
                Ct[mq * 64 + tm * 16 + half * 8 + v][col] = acc[tm][tn][v];
        }
    __syncthreads();

    if (z == 2) {
        // V epilogue: transposed bf16 store vt[h][d][s] (contiguous in s)
        int h2 = t >> 6, d = t & 63;                  // column t = head-half h2, dim d
        alignas(16) u16 tmp[128];
        for (int i = 0; i < 128; ++i) tmp[i] = f2bf(Ct[i][t]);
        u16* dst = vtb + ((size_t)(nt * 2 + h2) * HDIM + d) * SEQ + mBase;
        const v4u* tv = (const v4u*)tmp;
        v4u* dv = (v4u*)dst;
        for (int i = 0; i < 16; ++i) dv[i] = tv[i];
    } else {
        // Q/K epilogue: per-head RMSNorm (hd=64) + RoPE, bf16 store; thread t
        // owns row t of the tile, loops over the 2 heads in this N tile.
        const float* nw = (z == 0) ? qnw : knw;
        const int s = mBase + t;
        const float* cp = cosT + (size_t)s * 32;
        const float* sp = sinT + (size_t)s * 32;
        for (int hh = 0; hh < 2; ++hh) {
            float y[64];
            float ms = 0.f;
            for (int d = 0; d < 64; ++d) { float vq = Ct[t][hh * 64 + d]; y[d] = vq; ms += vq * vq; }
            float sc = rsqrtf(ms * (1.0f / 64.0f) + 1e-6f);
            for (int d = 0; d < 64; ++d) y[d] = y[d] * sc * nw[d];
            alignas(16) u16 outv[64];
            for (int j = 0; j < 32; ++j) {
                float c = cp[j], si = sp[j];
                float a = y[j], b = y[j + 32];
                outv[j]      = f2bf(a * c - b * si);
                outv[j + 32] = f2bf(b * c + a * si);
            }
            u16* dst = ((z == 0) ? qb : kb) + (size_t)s * DMODEL + (nt * 2 + hh) * 64;
            const v4u* ov = (const v4u*)outv;
            v4u* dv = (v4u*)dst;
            for (int i = 0; i < 8; ++i) dv[i] = ov[i];
        }
    }
}

// ---------------------------------------------------------------------------
// 2) Flash attention: block = (128 queries, 1 head), 128 threads = 4 waves.
//    Wave w owns query rows w*32..w*32+31 (2 row-subtiles) and all hd=64 cols.
//    Every thread owns one softmax row (m_i, l_i in registers).
// ---------------------------------------------------------------------------
__global__ __launch_bounds__(128) void flash_attn_kernel(
    const u16* __restrict__ qbuf,   // [S][D] bf16 (rms+rope applied)
    const u16* __restrict__ kbuf,   // [S][D] bf16
    const u16* __restrict__ vtb,    // [H][HD][S] bf16
    u16* __restrict__ obuf)         // [S][D] bf16
{
    const int q0 = blockIdx.x * 128;
    const int h  = blockIdx.y;

    __shared__ u16   Qt[128][72];   // [m][d]
    __shared__ u16   Kt[64][72];    // [key][d]
    __shared__ u16   Vt[64][72];    // [d][key]  (transposed V)
    __shared__ u16   Pt[128][72];   // [m][key]  softmax probs, bf16
    __shared__ float St[128][68];   // [m][key]  raw scores, fp32
    __shared__ float alpha_s[128];
    __shared__ float invl_s[128];

    const int t    = threadIdx.x;
    const int lane = t & 31;
    const int wave = t >> 5;
    const int half = lane >> 4;
    const int ln   = lane & 15;

    { // stage Q tile: thread t owns row t (64 bf16 = 8 x 16B)
        const u16* g = qbuf + (size_t)(q0 + t) * DMODEL + h * 64;
        u16*       l = &Qt[t][0];
        for (int c = 0; c < 64; c += 8) g2l_b128(g + c, l + c);
    }

    v8f o_acc[2][4];
    for (int i = 0; i < 2; ++i) for (int j = 0; j < 4; ++j)
        for (int e = 0; e < 8; ++e) o_acc[i][j][e] = 0.0f;
    float m_i = -1e30f, l_i = 0.0f;

    const int nKb = 2 * blockIdx.x + 2;     // causal: key blocks covering <= q0+127
    for (int kb = 0; kb < nKb; ++kb) {
        const int k0 = kb * 64;
        __syncthreads();                    // prev iter P/V reads complete
        { // K tile [key][d]: 2 threads per row, 32 bf16 each
            const u16* g = kbuf + (size_t)(k0 + (t >> 1)) * DMODEL + h * 64 + (t & 1) * 32;
            u16*       l = &Kt[t >> 1][(t & 1) * 32];
            for (int c = 0; c < 32; c += 8) g2l_b128(g + c, l + c);
        }
        { // V tile [d][key] from pre-transposed global (contiguous)
            const u16* g = vtb + ((size_t)h * HDIM + (t >> 1)) * SEQ + k0 + (t & 1) * 32;
            u16*       l = &Vt[t >> 1][(t & 1) * 32];
            for (int c = 0; c < 32; c += 8) g2l_b128(g + c, l + c);
        }
        g2l_fence();
        __syncthreads();

        // S = Q @ K^T : per wave 32 rows x 64 cols, reduce d (2 steps of 32)
        v8f s_acc[2][4];
        for (int i = 0; i < 2; ++i) for (int j = 0; j < 4; ++j)
            for (int e = 0; e < 8; ++e) s_acc[i][j][e] = 0.0f;
        for (int dk = 0; dk < 64; dk += 32) {
            Frag afr[2];
            for (int mt = 0; mt < 2; ++mt) {
                int m = wave * 32 + mt * 16 + ln;
                afr[mt].u[0] = *(const v4u*)&Qt[m][dk + half * 8];
                afr[mt].u[1] = *(const v4u*)&Qt[m][dk + 16 + half * 8];
            }
            for (int ntile = 0; ntile < 4; ++ntile) {
                Frag bfr;
                int n = ntile * 16 + ln;
                bfr.u[0] = *(const v4u*)&Kt[n][dk + half * 8];
                bfr.u[1] = *(const v4u*)&Kt[n][dk + 16 + half * 8];
                for (int mt = 0; mt < 2; ++mt)
                    s_acc[mt][ntile] = __builtin_amdgcn_wmma_f32_16x16x32_bf16(
                        false, afr[mt].v, false, bfr.v, (short)0, s_acc[mt][ntile], false, false);
            }
        }
        for (int mt = 0; mt < 2; ++mt)
            for (int ntile = 0; ntile < 4; ++ntile) {
                int col = ntile * 16 + ln;
                for (int v = 0; v < 8; ++v)
                    St[wave * 32 + mt * 16 + half * 8 + v][col] = s_acc[mt][ntile][v];
            }
        __syncthreads();

        // online softmax: thread t owns row t (all 128 threads active)
        {
            int grow = q0 + t;
            float sv[64];
            float mt2 = -1e30f;
            for (int j = 0; j < 64; ++j) {
                float sc = St[t][j] * 0.125f;         // 1/sqrt(64)
                if (k0 + j > grow) sc = -1e30f;       // causal mask
                sv[j] = sc;
                mt2 = fmaxf(mt2, sc);
            }
            float mnew = fmaxf(m_i, mt2);
            float al   = __expf(m_i - mnew);
            float ssum = 0.f;
            alignas(16) u16 prow[64];
            for (int j = 0; j < 64; ++j) {
                float p = __expf(sv[j] - mnew);
                ssum += p;
                prow[j] = f2bf(p);
            }
            l_i = l_i * al + ssum;
            m_i = mnew;
            alpha_s[t] = al;
            const v4u* pv = (const v4u*)prow;
            v4u* pd = (v4u*)&Pt[t][0];
            for (int i = 0; i < 8; ++i) pd[i] = pv[i];
        }
        __syncthreads();

        // rescale O by alpha, then O += P @ V
        {
            float al[2][8];
            for (int mt = 0; mt < 2; ++mt)
                for (int v = 0; v < 8; ++v)
                    al[mt][v] = alpha_s[wave * 32 + mt * 16 + half * 8 + v];
            for (int mt = 0; mt < 2; ++mt)
                for (int i = 0; i < 4; ++i)
                    for (int v = 0; v < 8; ++v) o_acc[mt][i][v] *= al[mt][v];
        }
        for (int kk = 0; kk < 64; kk += 32) {
            Frag afr[2];
            for (int mt = 0; mt < 2; ++mt) {
                int m = wave * 32 + mt * 16 + ln;
                afr[mt].u[0] = *(const v4u*)&Pt[m][kk + half * 8];
                afr[mt].u[1] = *(const v4u*)&Pt[m][kk + 16 + half * 8];
            }
            for (int ntile = 0; ntile < 4; ++ntile) {
                Frag bfr;
                int n = ntile * 16 + ln;              // d column
                bfr.u[0] = *(const v4u*)&Vt[n][kk + half * 8];
                bfr.u[1] = *(const v4u*)&Vt[n][kk + 16 + half * 8];
                for (int mt = 0; mt < 2; ++mt)
                    o_acc[mt][ntile] = __builtin_amdgcn_wmma_f32_16x16x32_bf16(
                        false, afr[mt].v, false, bfr.v, (short)0, o_acc[mt][ntile], false, false);
            }
        }
    }

    __syncthreads();
    invl_s[t] = 1.0f / l_i;
    __syncthreads();
    {
        float il[2][8];
        for (int mt = 0; mt < 2; ++mt)
            for (int v = 0; v < 8; ++v)
                il[mt][v] = invl_s[wave * 32 + mt * 16 + half * 8 + v];
        for (int mt = 0; mt < 2; ++mt)
            for (int ntile = 0; ntile < 4; ++ntile) {
                int col = h * 64 + ntile * 16 + ln;
                for (int v = 0; v < 8; ++v) {
                    int s = q0 + wave * 32 + mt * 16 + half * 8 + v;
                    obuf[(size_t)s * DMODEL + col] = f2bf(o_acc[mt][ntile][v] * il[mt][v]);
                }
            }
    }
}

// ---------------------------------------------------------------------------
// 3) Output projection: out = o @ wo^T, 128x128 tiles, fp32 straight to d_out.
// ---------------------------------------------------------------------------
__global__ __launch_bounds__(128) void out_proj_kernel(
    const u16* __restrict__ ob,   // [S][D] bf16
    const u16* __restrict__ wob,  // [N][K] bf16
    float* __restrict__ out)      // [S][D] fp32
{
    const int nBase = blockIdx.x * 128;
    const int mBase = blockIdx.y * 128;

    __shared__ u16 At[128][40];
    __shared__ u16 Bt[128][40];

    const int t    = threadIdx.x;
    const int lane = t & 31;
    const int wave = t >> 5;
    const int half = lane >> 4;
    const int ln   = lane & 15;
    const int mq   = wave >> 1, nq = wave & 1;

    v8f acc[4][4];
    for (int i = 0; i < 4; ++i) for (int j = 0; j < 4; ++j)
        for (int e = 0; e < 8; ++e) acc[i][j][e] = 0.0f;

    for (int k0 = 0; k0 < DMODEL; k0 += 32) {
        __syncthreads();
        {
            const u16* ga = ob + (size_t)(mBase + t) * DMODEL + k0;
            u16*       la = &At[t][0];
            const u16* gb = wob + (size_t)(nBase + t) * DMODEL + k0;
            u16*       lb = &Bt[t][0];
            for (int c = 0; c < 32; c += 8) g2l_b128(ga + c, la + c);
            for (int c = 0; c < 32; c += 8) g2l_b128(gb + c, lb + c);
            if (k0 + 32 < DMODEL) {
                __builtin_prefetch(ga + 32, 0, 3);
                __builtin_prefetch(gb + 32, 0, 3);
            }
        }
        g2l_fence();
        __syncthreads();

        Frag afr[4], bfr[4];
        for (int tm = 0; tm < 4; ++tm) {
            int m = mq * 64 + tm * 16 + ln;
            afr[tm].u[0] = *(const v4u*)&At[m][half * 8];
            afr[tm].u[1] = *(const v4u*)&At[m][16 + half * 8];
        }
        for (int tn = 0; tn < 4; ++tn) {
            int n = nq * 64 + tn * 16 + ln;
            bfr[tn].u[0] = *(const v4u*)&Bt[n][half * 8];
            bfr[tn].u[1] = *(const v4u*)&Bt[n][16 + half * 8];
        }
        for (int tm = 0; tm < 4; ++tm)
            for (int tn = 0; tn < 4; ++tn)
                acc[tm][tn] = __builtin_amdgcn_wmma_f32_16x16x32_bf16(
                    false, afr[tm].v, false, bfr[tn].v, (short)0, acc[tm][tn], false, false);
    }

    for (int tm = 0; tm < 4; ++tm)
        for (int tn = 0; tn < 4; ++tn) {
            int col = nBase + nq * 64 + tn * 16 + ln;
            for (int v = 0; v < 8; ++v) {
                int row = mBase + mq * 64 + tm * 16 + half * 8 + v;
                out[(size_t)row * DMODEL + col] = acc[tm][tn][v];
            }
        }
}

// ---------------------------------------------------------------------------
// launch
// ---------------------------------------------------------------------------
extern "C" void kernel_launch(void* const* d_in, const int* in_sizes, int n_in,
                              void* d_out, int out_size, void* d_ws, size_t ws_size,
                              hipStream_t stream) {
    const float* x    = (const float*)d_in[0];
    const float* cosT = (const float*)d_in[1];
    const float* sinT = (const float*)d_in[2];
    /* d_in[3] = attention_mask: causal, applied analytically (skip 64MB reads) */
    const float* wq   = (const float*)d_in[4];
    const float* wk   = (const float*)d_in[5];
    const float* wv   = (const float*)d_in[6];
    const float* wo   = (const float*)d_in[7];
    const float* qnw  = (const float*)d_in[8];
    const float* knw  = (const float*)d_in[9];
    float* out = (float*)d_out;

    char* ws = (char*)d_ws;
    const size_t MB = 1024 * 1024;
    u16* xb   = (u16*)(ws + 0);        // 8 MB  x bf16
    u16* wqb  = (u16*)(ws + 8  * MB);  // 2 MB
    u16* wkb  = (u16*)(ws + 10 * MB);  // 2 MB
    u16* wvb  = (u16*)(ws + 12 * MB);  // 2 MB
    u16* wob  = (u16*)(ws + 14 * MB);  // 2 MB
    u16* qbuf = (u16*)(ws + 16 * MB);  // 8 MB  Q (rms+rope) bf16
    u16* kbuf = (u16*)(ws + 24 * MB);  // 8 MB  K (rms+rope) bf16
    u16* vtb  = (u16*)(ws + 32 * MB);  // 8 MB  V transposed [H][HD][S]
    u16* obuf = (u16*)(ws + 40 * MB);  // 8 MB  attention output bf16

    cvt_bf16_kernel<<<(SEQ * DMODEL) / 1024, 256, 0, stream>>>(x,  xb,  SEQ * DMODEL);
    cvt_bf16_kernel<<<(DMODEL * DMODEL) / 1024, 256, 0, stream>>>(wq, wqb, DMODEL * DMODEL);
    cvt_bf16_kernel<<<(DMODEL * DMODEL) / 1024, 256, 0, stream>>>(wk, wkb, DMODEL * DMODEL);
    cvt_bf16_kernel<<<(DMODEL * DMODEL) / 1024, 256, 0, stream>>>(wv, wvb, DMODEL * DMODEL);
    cvt_bf16_kernel<<<(DMODEL * DMODEL) / 1024, 256, 0, stream>>>(wo, wob, DMODEL * DMODEL);

    qkv_gemm_kernel<<<dim3(DMODEL / 128, SEQ / 128, 3), 128, 0, stream>>>(
        xb, wqb, wkb, wvb, cosT, sinT, qnw, knw, qbuf, kbuf, vtb);

    flash_attn_kernel<<<dim3(SEQ / 128, NHEADC), 128, 0, stream>>>(qbuf, kbuf, vtb, obuf);

    out_proj_kernel<<<dim3(DMODEL / 128, SEQ / 128), 128, 0, stream>>>(obuf, wob, out);
}
